// Discriminator_AddDim_ESSAAttn_31026843746554
// MI455X (gfx1250) — compile-verified, matrix operand order checked by reading, not run on hardware
//
#include <hip/hip_runtime.h>

// ---------------------------------------------------------------------------
// Types / helpers
// ---------------------------------------------------------------------------
typedef __attribute__((ext_vector_type(16))) __bf16 v16bf;
typedef __attribute__((ext_vector_type(8)))  float  v8f;
typedef __attribute__((ext_vector_type(4)))  int    v4i;

union FragAB { v16bf v; unsigned int u[8]; };
union FragC  { v8f   v; float        f[8]; };

__device__ __forceinline__ float b2f(unsigned short h) {
  union { unsigned int u; float f; } x; x.u = ((unsigned int)h) << 16; return x.f;
}
__device__ __forceinline__ unsigned short f2b(float f) {
  union { float f; unsigned int u; } x; x.f = f;
  return (unsigned short)((x.u + 0x7FFFu + ((x.u >> 16) & 1u)) >> 16);
}

// ---------------------------------------------------------------------------
// CDNA5 async global->LDS copy (64 bytes = 4x b128), with sync fallback.
// Builtin signature (from compiler diagnostic): pointers are v4i* (generic).
// ---------------------------------------------------------------------------
#if defined(__gfx1250__) && __has_builtin(__builtin_amdgcn_global_load_async_to_lds_b128) && \
    __has_builtin(__builtin_amdgcn_s_wait_asynccnt)
#define USE_ASYNC_LDS 1
#else
#define USE_ASYNC_LDS 0
#endif

__device__ __forceinline__ void copy64B(const unsigned short* g, unsigned short* l) {
#if USE_ASYNC_LDS
  v4i* gp = (v4i*)g;
  v4i* lp = (v4i*)l;
  __builtin_amdgcn_global_load_async_to_lds_b128(gp + 0, lp + 0, 0, 0);
  __builtin_amdgcn_global_load_async_to_lds_b128(gp + 1, lp + 1, 0, 0);
  __builtin_amdgcn_global_load_async_to_lds_b128(gp + 2, lp + 2, 0, 0);
  __builtin_amdgcn_global_load_async_to_lds_b128(gp + 3, lp + 3, 0, 0);
#else
  const uint4* gs = (const uint4*)g;
  uint4* ls = (uint4*)l;
  ls[0] = gs[0]; ls[1] = gs[1]; ls[2] = gs[2]; ls[3] = gs[3];
#endif
}
__device__ __forceinline__ void wait_async() {
#if USE_ASYNC_LDS
  __builtin_amdgcn_s_wait_asynccnt(0);
#endif
}

// ---------------------------------------------------------------------------
// Generic batched bf16 WMMA GEMM:  C = act(alpha * A*B + bias + Res)
// A: MxK (KxM if TA), B: KxN (NxK if TB), bf16. Tile 64x64, K-stage 64
// (2 WMMA K-steps). 128 threads = 4 waves, each wave owns a 32x32 quadrant.
// act: 0=none 1=relu 2=leaky(0.01). Output bf16 (Co) or fp32 (Cf).
// ---------------------------------------------------------------------------
template <int TA, int TB>
__global__ __launch_bounds__(128) void gemm_bf16_kernel(
    const unsigned short* __restrict__ A, long lda, long sA,
    const unsigned short* __restrict__ Bm, long ldb, long sB,
    const float* __restrict__ bias,
    const unsigned short* __restrict__ Res, long ldres, long sR,
    unsigned short* __restrict__ Co, float* __restrict__ Cf, long ldc, long sC,
    int M, int N, int K, float alpha, int act)
{
  __shared__ unsigned short As[64][64];
  __shared__ unsigned short Bs[64][64];

  const int tid = threadIdx.x;
  const int m0  = blockIdx.x * 64;
  const int n0  = blockIdx.y * 64;
  const long z  = blockIdx.z;
  A += z * sA; Bm += z * sB;
  if (Res) Res += z * sR;
  if (Co)  Co  += z * sC;
  if (Cf)  Cf  += z * sC;

  const int lane = tid & 31;
  const int wave = tid >> 5;
  const int wm   = (wave >> 1) * 32;
  const int wn   = (wave & 1) * 32;
  const int lrow = lane & 15;
  const int kw   = (lane >> 4) * 4;     // u32 offset in 32-elem bf16 K-slice

  const bool fullM = (m0 + 64 <= M);
  const bool fullN = (n0 + 64 <= N);
  const int  crow  = tid >> 1;          // staging row for fast path
  const int  ck0   = (tid & 1) * 32;    // staging K-offset for fast path

  v8f acc[2][2] = {};

  for (int kb = 0; kb < K; kb += 64) {
    const bool fullK = (kb + 64 <= K);

    // ---- stage A tile -> As[m][k] ----
    if (TA == 0) {
      if (fullM && fullK) {
        copy64B(A + (long)(m0 + crow) * lda + kb + ck0, &As[crow][ck0]);
      } else {
        for (int e = tid; e < 4096; e += 128) {
          int m = e >> 6, k = e & 63;
          int gm = m0 + m, gk = kb + k;
          int cm = gm < M ? gm : M - 1;
          int cc = gk < K ? gk : K - 1;
          unsigned short v = A[(long)cm * lda + cc];
          As[m][k] = (gm < M && gk < K) ? v : (unsigned short)0;
        }
      }
    } else {
      for (int e = tid; e < 4096; e += 128) {   // A is KxM
        int k = e >> 6, m = e & 63;
        int gm = m0 + m, gk = kb + k;
        int cm = gm < M ? gm : M - 1;
        int cc = gk < K ? gk : K - 1;
        unsigned short v = A[(long)cc * lda + cm];
        As[m][k] = (gm < M && gk < K) ? v : (unsigned short)0;
      }
    }

    // ---- stage B tile -> Bs[n][k] (K-major per n) ----
    if (TB == 1) {
      if (fullN && fullK) {
        copy64B(Bm + (long)(n0 + crow) * ldb + kb + ck0, &Bs[crow][ck0]);
      } else {
        for (int e = tid; e < 4096; e += 128) {
          int n = e >> 6, k = e & 63;
          int gn = n0 + n, gk = kb + k;
          int cn = gn < N ? gn : N - 1;
          int cc = gk < K ? gk : K - 1;
          unsigned short v = Bm[(long)cn * ldb + cc];
          Bs[n][k] = (gn < N && gk < K) ? v : (unsigned short)0;
        }
      }
    } else {
      for (int e = tid; e < 4096; e += 128) {   // B is KxN
        int k = e >> 6, n = e & 63;
        int gn = n0 + n, gk = kb + k;
        int cn = gn < N ? gn : N - 1;
        int cc = gk < K ? gk : K - 1;
        unsigned short v = Bm[(long)cc * ldb + cn];
        Bs[n][k] = (gn < N && gk < K) ? v : (unsigned short)0;
      }
    }

    wait_async();
    __syncthreads();

#pragma unroll
    for (int s = 0; s < 2; ++s) {
      FragAB fa[2], fb[2];
#pragma unroll
      for (int i = 0; i < 2; ++i) {
        const unsigned int* rp = (const unsigned int*)&As[wm + i * 16 + lrow][s * 32];
#pragma unroll
        for (int j = 0; j < 4; ++j) { fa[i].u[j] = rp[kw + j]; fa[i].u[4 + j] = rp[8 + kw + j]; }
      }
#pragma unroll
      for (int i = 0; i < 2; ++i) {
        const unsigned int* rp = (const unsigned int*)&Bs[wn + i * 16 + lrow][s * 32];
#pragma unroll
        for (int j = 0; j < 4; ++j) { fb[i].u[j] = rp[kw + j]; fb[i].u[4 + j] = rp[8 + kw + j]; }
      }
#pragma unroll
      for (int i = 0; i < 2; ++i)
#pragma unroll
        for (int j = 0; j < 2; ++j)
          acc[i][j] = __builtin_amdgcn_wmma_f32_16x16x32_bf16(
              false, fa[i].v, false, fb[j].v, (short)0, acc[i][j], false, false);
    }
    __syncthreads();
  }

  // ---- epilogue (C 16x16 f32 layout: VGPR e -> M = e + 8*(lane>>4), N = lane&15) ----
#pragma unroll
  for (int i = 0; i < 2; ++i) {
#pragma unroll
    for (int j = 0; j < 2; ++j) {
      FragC c; c.v = acc[i][j];
      int col   = n0 + wn + j * 16 + lrow;
      int rbase = m0 + wm + i * 16 + 8 * (lane >> 4);
#pragma unroll
      for (int e = 0; e < 8; ++e) {
        int row = rbase + e;
        if (row < M && col < N) {
          float v = c.f[e] * alpha;
          if (bias) v += bias[col];
          if (Res)  v += b2f(Res[(long)row * ldres + col]);
          if (act == 1)      v = fmaxf(v, 0.f);
          else if (act == 2) v = v > 0.f ? v : 0.01f * v;
          if (Co) Co[(long)row * ldc + col] = f2b(v);
          else    Cf[(long)row * ldc + col] = v;
        }
      }
    }
  }
}

// ---------------------------------------------------------------------------
// Implicit-im2col 3x3 VALID conv as WMMA GEMM.
// M = B*OH*OW rows, K = IC*9 (multiple of 64 here), N = OC (multiple of 64).
// Input NCHW bf16, weights bf16 flat [OC][IC*9], output NCHW bf16, bias+relu.
// ---------------------------------------------------------------------------
__global__ __launch_bounds__(128) void conv3x3_wmma_kernel(
    const unsigned short* __restrict__ In, const unsigned short* __restrict__ Wt,
    const float* __restrict__ bias, unsigned short* __restrict__ Out,
    int IC, int IH, int IW, int OC, int OH, int OW, int M)
{
  __shared__ unsigned short As[64][64];
  __shared__ unsigned short Bs[64][64];

  const int K    = IC * 9;
  const int tid  = threadIdx.x;
  const int m0   = blockIdx.x * 64;
  const int n0   = blockIdx.y * 64;
  const int lane = tid & 31;
  const int wave = tid >> 5;
  const int wm   = (wave >> 1) * 32;
  const int wn   = (wave & 1) * 32;
  const int lrow = lane & 15;
  const int kw   = (lane >> 4) * 4;
  const int hw   = OH * OW;
  const int crow = tid >> 1;
  const int ck0  = (tid & 1) * 32;

  v8f acc[2][2] = {};

  for (int kb = 0; kb < K; kb += 64) {
    // implicit im2col A tile (branch-free: clamp + select)
    for (int e = tid; e < 4096; e += 128) {
      int m = e >> 6, k = e & 63;
      int gm = m0 + m, gk = kb + k;
      int cm = gm < M ? gm : M - 1;
      int b   = cm / hw;  int rem = cm - b * hw;
      int oy  = rem / OW; int ox  = rem - oy * OW;
      int ic  = gk / 9;   int t9  = gk - ic * 9;
      int ky  = t9 / 3;   int kx  = t9 - ky * 3;
      unsigned short v = In[(((long)(b * IC + ic)) * IH + (oy + ky)) * IW + (ox + kx)];
      As[m][k] = (gm < M) ? v : (unsigned short)0;
    }
    // weights [OC][K] -> Bs[n][k]: contiguous, async
    copy64B(Wt + (long)(n0 + crow) * K + kb + ck0, &Bs[crow][ck0]);

    wait_async();
    __syncthreads();

#pragma unroll
    for (int s = 0; s < 2; ++s) {
      FragAB fa[2], fb[2];
#pragma unroll
      for (int i = 0; i < 2; ++i) {
        const unsigned int* rp = (const unsigned int*)&As[wm + i * 16 + lrow][s * 32];
#pragma unroll
        for (int j = 0; j < 4; ++j) { fa[i].u[j] = rp[kw + j]; fa[i].u[4 + j] = rp[8 + kw + j]; }
      }
#pragma unroll
      for (int i = 0; i < 2; ++i) {
        const unsigned int* rp = (const unsigned int*)&Bs[wn + i * 16 + lrow][s * 32];
#pragma unroll
        for (int j = 0; j < 4; ++j) { fb[i].u[j] = rp[kw + j]; fb[i].u[4 + j] = rp[8 + kw + j]; }
      }
#pragma unroll
      for (int i = 0; i < 2; ++i)
#pragma unroll
        for (int j = 0; j < 2; ++j)
          acc[i][j] = __builtin_amdgcn_wmma_f32_16x16x32_bf16(
              false, fa[i].v, false, fb[j].v, (short)0, acc[i][j], false, false);
    }
    __syncthreads();
  }

#pragma unroll
  for (int i = 0; i < 2; ++i) {
#pragma unroll
    for (int j = 0; j < 2; ++j) {
      FragC c; c.v = acc[i][j];
      int oc    = n0 + wn + j * 16 + lrow;
      int rbase = m0 + wm + i * 16 + 8 * (lane >> 4);
#pragma unroll
      for (int e = 0; e < 8; ++e) {
        int row = rbase + e;
        if (row < M) {
          int b  = row / hw;  int rem = row - b * hw;
          int oy = rem / OW;  int ox  = rem - oy * OW;
          float v = fmaxf(c.f[e] + bias[oc], 0.f);   // bias + relu
          Out[(((long)(b * OC + oc)) * OH + oy) * OW + ox] = f2b(v);
        }
      }
    }
  }
}

// ---------------------------------------------------------------------------
// Elementwise / reduction helpers
// ---------------------------------------------------------------------------
__global__ void f32_to_bf16_kernel(const float* __restrict__ s,
                                   unsigned short* __restrict__ d, long n) {
  long i = (long)blockIdx.x * blockDim.x + threadIdx.x;
  if (i < n) d[i] = f2b(s[i]);
}

// x [B,C,HW] fp32 -> t [B*HW, C] bf16
__global__ void nchw_to_tok_kernel(const float* __restrict__ x,
                                   unsigned short* __restrict__ t,
                                   long Bn, long Cn, long HW) {
  long i = (long)blockIdx.x * blockDim.x + threadIdx.x;
  long n = Bn * Cn * HW;
  if (i >= n) return;
  long b = i / (Cn * HW); long rem = i - b * Cn * HW;
  long c = rem / HW;      long p = rem - c * HW;
  t[(b * HW + p) * Cn + c] = f2b(x[i]);
}

// t [B*HW, C] bf16 -> x [B,C,HW] bf16
__global__ void tok_to_nchw_kernel(const unsigned short* __restrict__ t,
                                   unsigned short* __restrict__ x,
                                   long Bn, long Cn, long HW) {
  long i = (long)blockIdx.x * blockDim.x + threadIdx.x;
  long n = Bn * Cn * HW;
  if (i >= n) return;
  long b = i / (Cn * HW); long rem = i - b * Cn * HW;
  long c = rem / HW;      long p = rem - c * HW;
  x[i] = t[(b * HW + p) * Cn + c];
}

__device__ __forceinline__ float blockSum128(float v, float* red) {
  int t = threadIdx.x;
  red[t] = v; __syncthreads();
  for (int s = 64; s > 0; s >>= 1) { if (t < s) red[t] += red[t + s]; __syncthreads(); }
  float r = red[0]; __syncthreads();
  return r;
}

// Per-token ESSA normalization: one block (128 thr) per token.
__global__ __launch_bounds__(128) void essa_norm_kernel(
    const unsigned short* __restrict__ qkv,
    unsigned short* __restrict__ q2o, unsigned short* __restrict__ k2o) {
  __shared__ float red[128];
  long row = blockIdx.x;
  int  c   = threadIdx.x;
  float q = b2f(qkv[row * 384 + c]);
  float k = b2f(qkv[row * 384 + 128 + c]);
  float mq = blockSum128(q, red) * (1.f / 128.f);
  float mk = blockSum128(k, red) * (1.f / 128.f);
  float qc = q - mq, kc = k - mk;
  float q2 = qc * qc, k2 = kc * kc;
  float sq = blockSum128(q2, red);
  float sk = blockSum128(k2, red);
  q2 = q2 / (sq + 1e-7f);
  k2 = k2 / (sk + 1e-7f);
  float l2 = sqrtf(blockSum128(q2 * q2, red));
  q2 = q2 / fmaxf(l2, 1e-12f);
  q2o[row * 128 + c] = f2b(q2);
  k2o[row * 128 + c] = f2b(k2);
}

// Per-(batch,channel) L2 norm of k2 over tokens: 1 block/batch, 128 threads.
__global__ __launch_bounds__(128) void k2_colnorm_kernel(
    const unsigned short* __restrict__ k2, float* __restrict__ norm, int NT) {
  int b = blockIdx.x, c = threadIdx.x;
  const unsigned short* p = k2 + (long)b * NT * 128 + c;
  float s = 0.f;
  for (int n = 0; n < NT; ++n) { float v = b2f(p[(long)n * 128]); s += v * v; }
  norm[b * 128 + c] = fmaxf(sqrtf(s), 1e-12f);
}

__global__ void k2_scale_kernel(unsigned short* __restrict__ k2,
                                const float* __restrict__ norm, long total, int NT) {
  long i = (long)blockIdx.x * blockDim.x + threadIdx.x;
  if (i >= total) return;
  long row = i >> 7; int c = (int)(i & 127);
  int b = (int)(row / NT);
  k2[i] = f2b(b2f(k2[i]) / norm[b * 128 + c]);
}

__global__ void maxpool2_kernel(const unsigned short* __restrict__ in,
                                unsigned short* __restrict__ out,
                                int C, int H, int W, int OH, int OW, long total) {
  long i = (long)blockIdx.x * blockDim.x + threadIdx.x;
  if (i >= total) return;
  int ox = (int)(i % OW); long r = i / OW;
  int oy = (int)(r % OH); r /= OH;
  int c  = (int)(r % C);  int b = (int)(r / C);
  const unsigned short* p = in + (((long)(b * C + c)) * H + oy * 2) * W + ox * 2;
  float m = b2f(p[0]);
  m = fmaxf(m, b2f(p[1]));
  m = fmaxf(m, b2f(p[W]));
  m = fmaxf(m, b2f(p[W + 1]));
  out[i] = f2b(m);
}

// ---------------------------------------------------------------------------
// Host-side orchestration
// ---------------------------------------------------------------------------
extern "C" void kernel_launch(void* const* d_in, const int* in_sizes, int n_in,
                              void* d_out, int out_size, void* d_ws, size_t ws_size,
                              hipStream_t stream) {
  (void)in_sizes; (void)n_in; (void)out_size; (void)ws_size;

  const float* x      = (const float*)d_in[0];
  const float* w_qkv  = (const float*)d_in[1];
  const float* b_qkv  = (const float*)d_in[2];
  const float* w_ln   = (const float*)d_in[3];
  const float* b_ln   = (const float*)d_in[4];
  const float* w_ffn1 = (const float*)d_in[5];
  const float* b_ffn1 = (const float*)d_in[6];
  const float* w_ffn2 = (const float*)d_in[7];
  const float* b_ffn2 = (const float*)d_in[8];
  const float* w_c1   = (const float*)d_in[9];
  const float* b_c1   = (const float*)d_in[10];
  const float* w_c2   = (const float*)d_in[11];
  const float* b_c2   = (const float*)d_in[12];
  const float* w_fc1  = (const float*)d_in[13];
  const float* b_fc1  = (const float*)d_in[14];
  const float* w_fc2  = (const float*)d_in[15];
  const float* b_fc2  = (const float*)d_in[16];
  const float* w_cls  = (const float*)d_in[17];
  const float* b_cls  = (const float*)d_in[18];

  const long Bn = 256, Cn = 128, HW = 729, NT = 729;
  const long T = Bn * HW;                  // 186624 tokens

  char* base = (char*)d_ws;
  size_t off = 0;
  auto alloc = [&](size_t elems, size_t esz) -> size_t {
    size_t o = off; off += ((elems * esz + 255) & ~(size_t)255); return o;
  };
  // bf16 activation buffers
  size_t o_t    = alloc((size_t)T * 128, 2);       // tokens; later: conv stack
  size_t o_qkv  = alloc((size_t)T * 384, 2);       // later: h_tmp/h/x_en_t/x_en
  size_t o_q2   = alloc((size_t)T * 128, 2);       // later: attn
  size_t o_k2   = alloc((size_t)T * 128, 2);       // later: attn_in
  size_t o_kv   = alloc((size_t)Bn * 128 * 128, 2);
  size_t o_norm = alloc((size_t)Bn * 128, 4);      // fp32
  // bf16 weights
  size_t o_wqkv = alloc(128 * 384, 2);
  size_t o_wln  = alloc(128 * 128, 2);
  size_t o_wf1  = alloc(64 * 256, 2);
  size_t o_wf2  = alloc(128 * 64, 2);
  size_t o_wc1  = alloc(64 * 128 * 9, 2);
  size_t o_wc2  = alloc(128 * 64 * 9, 2);
  size_t o_wfc1 = alloc(3200 * 512, 2);
  size_t o_wfc2 = alloc(512 * 512, 2);
  size_t o_wcls = alloc(512 * 16, 2);
  // aliases
  size_t o_attn_in = o_k2;
  size_t o_attn    = o_q2;
  size_t o_htmp    = o_qkv;
  size_t o_h       = o_qkv + (size_t)T * 64 * 2;
  size_t o_xent    = o_qkv + (size_t)T * 128 * 2;
  size_t o_xen     = o_qkv + (size_t)T * 256 * 2;
  size_t o_c1      = o_t;
  size_t o_p1      = o_t  + (size_t)10240000 * 2;
  size_t o_c2      = o_p1 + (size_t)2359296 * 2;
  size_t o_p2      = o_c2 + (size_t)3276800 * 2;
  size_t o_o3      = o_p2 + (size_t)819200 * 2;
  size_t o_o4      = o_o3 + (size_t)131072 * 2;

  auto U = [&](size_t o) { return (unsigned short*)(base + o); };
  auto F = [&](size_t o) { return (float*)(base + o); };
  auto cvt = [&](const float* s, size_t o, long n) {
    f32_to_bf16_kernel<<<dim3((unsigned)((n + 255) / 256)), dim3(256), 0, stream>>>(s, U(o), n);
  };

  // ---- weight conversion fp32 -> bf16 ----
  cvt(w_qkv, o_wqkv, 128 * 384);
  cvt(w_ln,  o_wln,  128 * 128);
  cvt(w_ffn1, o_wf1, 64 * 256);
  cvt(w_ffn2, o_wf2, 128 * 64);
  cvt(w_c1,  o_wc1,  64 * 128 * 9);
  cvt(w_c2,  o_wc2,  128 * 64 * 9);
  cvt(w_fc1, o_wfc1, 3200 * 512);
  cvt(w_fc2, o_wfc2, 512 * 512);
  cvt(w_cls, o_wcls, 512 * 16);

  const dim3 blk(128);
  const unsigned short* NUS = nullptr;
  float* NF = nullptr;

  // ---- x NCHW -> tokens ----
  {
    long n = Bn * Cn * HW;
    nchw_to_tok_kernel<<<dim3((unsigned)((n + 255) / 256)), dim3(256), 0, stream>>>(x, U(o_t), Bn, Cn, HW);
  }

  // ---- qkv = t @ w_qkv + b_qkv   [T,128]x[128,384] ----
  gemm_bf16_kernel<0,0><<<dim3(2916, 6, 1), blk, 0, stream>>>(
      U(o_t), 128, 0, U(o_wqkv), 384, 0, b_qkv, NUS, 0, 0,
      U(o_qkv), NF, 384, 0, (int)T, 384, 128, 1.f, 0);

  // ---- per-token ESSA normalization ----
  essa_norm_kernel<<<dim3((unsigned)T), blk, 0, stream>>>(U(o_qkv), U(o_q2), U(o_k2));

  // ---- k2 column (token-dim) L2 norm + scale ----
  k2_colnorm_kernel<<<dim3((unsigned)Bn), blk, 0, stream>>>(U(o_k2), F(o_norm), (int)NT);
  {
    long n = T * 128;
    k2_scale_kernel<<<dim3((unsigned)((n + 255) / 256)), dim3(256), 0, stream>>>(U(o_k2), F(o_norm), n, (int)NT);
  }

  // ---- kv[b] = k2[b]^T @ v[b]   (M=128,N=128,K=729, batched 256) ----
  gemm_bf16_kernel<1,0><<<dim3(2, 2, 256), blk, 0, stream>>>(
      U(o_k2), 128, NT * 128, U(o_qkv) + 256, 384, NT * 384, nullptr, NUS, 0, 0,
      U(o_kv), NF, 128, 128 * 128, 128, 128, (int)NT, 1.f, 0);

  // ---- attn_in[b] = v[b] + (q2[b] @ kv[b]) / 27 ----
  gemm_bf16_kernel<0,0><<<dim3(12, 2, 256), blk, 0, stream>>>(
      U(o_q2), 128, NT * 128, U(o_kv), 128, 128 * 128, nullptr,
      U(o_qkv) + 256, 384, NT * 384,
      U(o_attn_in), NF, 128, NT * 128, (int)NT, 128, 128, 1.f / 27.f, 0);

  // ---- attn = attn_in @ w_ln + b_ln ----
  gemm_bf16_kernel<0,0><<<dim3(2916, 2, 1), blk, 0, stream>>>(
      U(o_attn_in), 128, 0, U(o_wln), 128, 0, b_ln, NUS, 0, 0,
      U(o_attn), NF, 128, 0, (int)T, 128, 128, 1.f, 0);

  // ---- FFN 1x1 convs in token space (cat split into two accumulating GEMMs) ----
  gemm_bf16_kernel<0,1><<<dim3(2916, 1, 1), blk, 0, stream>>>(     // h_tmp = t @ w1[:, :128]^T
      U(o_t), 128, 0, U(o_wf1), 256, 0, nullptr, NUS, 0, 0,
      U(o_htmp), NF, 64, 0, (int)T, 64, 128, 1.f, 0);
  gemm_bf16_kernel<0,1><<<dim3(2916, 1, 1), blk, 0, stream>>>(     // h = leaky(h_tmp + attn@w1[:,128:]^T + b)
      U(o_attn), 128, 0, U(o_wf1) + 128, 256, 0, b_ffn1,
      U(o_htmp), 64, 0,
      U(o_h), NF, 64, 0, (int)T, 64, 128, 1.f, 2);
  gemm_bf16_kernel<0,1><<<dim3(2916, 2, 1), blk, 0, stream>>>(     // x_en_t = h @ w2^T + b + t
      U(o_h), 64, 0, U(o_wf2), 64, 0, b_ffn2,
      U(o_t), 128, 0,
      U(o_xent), NF, 128, 0, (int)T, 128, 64, 1.f, 0);

  // ---- tokens -> NCHW ----
  {
    long n = Bn * Cn * HW;
    tok_to_nchw_kernel<<<dim3((unsigned)((n + 255) / 256)), dim3(256), 0, stream>>>(U(o_xent), U(o_xen), Bn, Cn, HW);
  }

  // ---- conv1: 128->64, 27x27 -> 25x25, relu ----
  conv3x3_wmma_kernel<<<dim3(2500, 1, 1), blk, 0, stream>>>(
      U(o_xen), U(o_wc1), b_c1, U(o_c1), 128, 27, 27, 64, 25, 25, 256 * 625);
  {
    long n = 256L * 64 * 12 * 12;
    maxpool2_kernel<<<dim3((unsigned)((n + 255) / 256)), dim3(256), 0, stream>>>(
        U(o_c1), U(o_p1), 64, 25, 25, 12, 12, n);
  }

  // ---- conv2: 64->128, 12x12 -> 10x10, relu ----
  conv3x3_wmma_kernel<<<dim3(400, 2, 1), blk, 0, stream>>>(
      U(o_p1), U(o_wc2), b_c2, U(o_c2), 64, 12, 12, 128, 10, 10, 256 * 100);
  {
    long n = 256L * 128 * 5 * 5;
    maxpool2_kernel<<<dim3((unsigned)((n + 255) / 256)), dim3(256), 0, stream>>>(
        U(o_c2), U(o_p2), 128, 10, 10, 5, 5, n);
  }

  // ---- FC stack ----
  gemm_bf16_kernel<0,0><<<dim3(4, 8, 1), blk, 0, stream>>>(        // o3 = relu(o2 @ w_fc1 + b)
      U(o_p2), 3200, 0, U(o_wfc1), 512, 0, b_fc1, NUS, 0, 0,
      U(o_o3), NF, 512, 0, 256, 512, 3200, 1.f, 1);
  gemm_bf16_kernel<0,0><<<dim3(4, 8, 1), blk, 0, stream>>>(        // o4 = relu(o3 @ w_fc2 + b)
      U(o_o3), 512, 0, U(o_wfc2), 512, 0, b_fc2, NUS, 0, 0,
      U(o_o4), NF, 512, 0, 256, 512, 512, 1.f, 1);
  gemm_bf16_kernel<0,0><<<dim3(4, 1, 1), blk, 0, stream>>>(        // out = o4 @ w_cls + b  (fp32 out)
      U(o_o4), 512, 0, U(o_wcls), 16, 0, b_cls, NUS, 0, 0,
      (unsigned short*)nullptr, (float*)d_out, 16, 0, 256, 16, 512, 1.f, 0);
}
